// SoftRedProcessor_62105227100549
// MI455X (gfx1250) — matrix operand order repaired and checked by reading, not compiled, not run on hardware
//
#include <hip/hip_runtime.h>
#include <cstdint>

#define VOCAB     50257
#define BATCH     512
#define HHALF     25129            // ceil(VOCAB/2), JAX threefry half-split
#define GWORDS    1571             // ceil(VOCAB/32)
#define NBINS     256
#define HIST_LO   (-10.0f)
#define HIST_INVW (256.0f / 20.0f)
#define HIST_W    (20.0f / 256.0f)
#define EXP_SHIFT 6.0f
#define BIASF     2.0f
#define TOPP      0.9f
#define TILE_ELEMS 1024            // 256 threads * 4 floats (b128)

typedef __attribute__((ext_vector_type(16))) _Float16 v16h;
typedef __attribute__((ext_vector_type(8)))  float    v8f;

struct U2 { uint32_t x, y; };

__device__ __forceinline__ uint32_t rotl32(uint32_t v, int r) { return (v << r) | (v >> (32 - r)); }

// threefry2x32-20, matching JAX's PRNG core
__device__ __forceinline__ U2 threefry2x32(uint32_t k0, uint32_t k1, uint32_t x0, uint32_t x1) {
  uint32_t k2 = k0 ^ k1 ^ 0x1BD11BDAu;
  x0 += k0; x1 += k1;
#define TFR(r) { x0 += x1; x1 = rotl32(x1, r); x1 ^= x0; }
  TFR(13) TFR(15) TFR(26) TFR(6)  x0 += k1; x1 += k2 + 1u;
  TFR(17) TFR(29) TFR(16) TFR(24) x0 += k2; x1 += k0 + 2u;
  TFR(13) TFR(15) TFR(26) TFR(6)  x0 += k0; x1 += k1 + 3u;
  TFR(17) TFR(29) TFR(16) TFR(24) x0 += k1; x1 += k2 + 4u;
  TFR(13) TFR(15) TFR(26) TFR(6)  x0 += k2; x1 += k0 + 5u;
#undef TFR
  return {x0, x1};
}

// low 32 bits of a flat pointer into __shared__ == LDS byte address (ISA 10.2 aperture calc)
__device__ __forceinline__ uint32_t lds_off32(const void* p) {
  return (uint32_t)(uintptr_t)p;
}

// CDNA5 async data mover: global -> LDS, tracked by ASYNCcnt (CDNA5 ISA)
__device__ __forceinline__ void async_load_b128(uint32_t lds_dst, const float* gsrc) {
  asm volatile("global_load_async_to_lds_b128 %0, %1, off"
               :
               : "v"(lds_dst), "v"(gsrc)
               : "memory");
}
__device__ __forceinline__ void wait_async0() { asm volatile("s_wait_asynccnt 0x0" ::: "memory"); }
__device__ __forceinline__ void wait_async1() { asm volatile("s_wait_asynccnt 0x1" ::: "memory"); }

__global__ __launch_bounds__(256) void wm_topp_kernel(const int* __restrict__ ids,
                                                      const float* __restrict__ scores,
                                                      int* __restrict__ tok) {
  __shared__ __align__(16) float tile[2][TILE_ELEMS];
  __shared__ float    hist[NBINS];
  __shared__ uint32_t green[GWORDS];
  __shared__ float    scanbuf[256];
  __shared__ float    bestv[256];
  __shared__ int      besti[256];
  __shared__ float    sh_thresh;

  const int tid = threadIdx.x;
  const int row = blockIdx.x;
  const float* base = scores + (size_t)row * VOCAB;

  // ---- per-row keys (redundant per thread; cheap) ----
  int ss = ids[row * 3 + 0] + ids[row * 3 + 1] + ids[row * 3 + 2];
  uint32_t d = ((uint32_t)(42u * (uint32_t)ss)) % 2147483647u;
  U2 rk = threefry2x32(0u, 12345u, 0u, d);       // fold_in(key(12345), d)
  U2 gk = threefry2x32(rk.x, rk.y, 0u, 0u);      // fold_in(rk, 0)
  U2 sk = threefry2x32(rk.x, rk.y, 0u, 1u);      // fold_in(rk, 1)

  // ---- init LDS ----
  if (tid < NBINS) hist[tid] = 0.0f;
  for (int w = tid; w < GWORDS; w += 256) green[w] = 0u;
  __syncthreads();

  // ---- phase G: green bitmask via threefry half-split counters ----
  for (int j = tid; j < HHALF; j += 256) {
    U2 b = threefry2x32(gk.x, gk.y, (uint32_t)j, (uint32_t)(j + HHALF));
    if (!(b.x >> 31)) atomicOr(&green[j >> 5], 1u << (j & 31));     // uniform<0.5 <=> top bit 0
    int v2 = j + HHALF;
    if (v2 < VOCAB && !(b.y >> 31)) atomicOr(&green[v2 >> 5], 1u << (v2 & 31));
  }
  __syncthreads();

  const int NT = (VOCAB + TILE_ELEMS - 1) / TILE_ELEMS;  // 50

  // ---- pass AB: stream scores via async-to-LDS, build prob-mass histogram ----
  {
    int off0 = tid * 4;
    int c0 = min(off0, VOCAB - 4);
    async_load_b128(lds_off32(&tile[0][tid * 4]), base + c0);
    for (int t = 0; t < NT; ++t) {
      if (t + 1 < NT) {
        int offn = (t + 1) * TILE_ELEMS + tid * 4;
        int cn = min(offn, VOCAB - 4);
        async_load_b128(lds_off32(&tile[(t + 1) & 1][tid * 4]), base + cn);
        wait_async1();
      } else {
        wait_async0();
      }
      int off = t * TILE_ELEMS + tid * 4;
      int c = min(off, VOCAB - 4);
#pragma unroll
      for (int k = 0; k < 4; ++k) {
        int v = c + k;
        if (v >= off && v < VOCAB) {
          float sv = tile[t & 1][tid * 4 + k];
          float gb = ((green[v >> 5] >> (v & 31)) & 1u) ? BIASF : 0.0f;
          float l = sv + gb;
          int b = (int)((l - HIST_LO) * HIST_INVW);
          b = b < 0 ? 0 : (b > 255 ? 255 : b);
          atomicAdd(&hist[b], __expf(l - EXP_SHIFT));
        }
      }
    }
  }
  __syncthreads();

  // ---- WMMA suffix-scan of histogram: D = A(rev bins, 16x32 f16) x B(lower-tri ones) ----
  if (tid < 32) {
    const int lane = tid;
    const int M = lane & 15;
    const bool hi = lane >= 16;
    v16h a, bm;
#pragma unroll
    for (int e = 0; e < 16; ++e) {
      // A layout (16-bit 16x32, ISA 7.12.2): M = lane&15; K = e + (e>=8?8:0) + (lane>=16?8:0)
      int K = e + ((e >= 8) ? 8 : 0) + (hi ? 8 : 0);
      float av = (K < 16) ? hist[255 - (16 * M + K)] : 0.0f;
      a[e] = (_Float16)av;
      // B layout (16-bit 32x16): N = lane&15; K = same as A; B[k][n]=(k<=n && k<16)
      float bv = (K < 16 && K <= M) ? 1.0f : 0.0f;
      bm[e] = (_Float16)bv;
    }
    v8f cacc = {0.f, 0.f, 0.f, 0.f, 0.f, 0.f, 0.f, 0.f};
    cacc = __builtin_amdgcn_wmma_f32_16x16x32_f16(false, a, false, bm, (short)0, cacc,
                                                  false, false);
#pragma unroll
    for (int j = 0; j < 8; ++j) {
      // C/D layout (32-bit 16x16): M = j + (lane>=16?8:0); N = lane&15
      int m = j + (hi ? 8 : 0);
      scanbuf[m * 16 + (lane & 15)] = cacc[j];
    }
  }
  __syncthreads();

  // ---- carry fixup + top-p cutoff (256 serial steps, negligible) ----
  if (tid == 0) {
    float Z = 0.0f;
    for (int m = 0; m < 16; ++m) Z += scanbuf[m * 16 + 15];
    float target = TOPP * Z;
    float carry = 0.0f;
    int istar = 255;
    bool found = false;
    for (int m = 0; m < 16 && !found; ++m) {
      for (int n = 0; n < 16; ++n) {
        if (carry + scanbuf[m * 16 + n] >= target) { istar = m * 16 + n; found = true; break; }
      }
      if (!found) carry += scanbuf[m * 16 + 15];
    }
    int bstar = 255 - istar;
    sh_thresh = HIST_LO + (float)bstar * HIST_W;   // keep logits >= lower edge of cutoff bin
  }
  __syncthreads();

  // ---- pass C: Gumbel-max categorical sample over kept set ----
  float bv = -3.4e38f;
  int bi = 0;
  const float thresh = sh_thresh;
  {
    int off0 = tid * 4;
    int c0 = min(off0, VOCAB - 4);
    async_load_b128(lds_off32(&tile[0][tid * 4]), base + c0);
    for (int t = 0; t < NT; ++t) {
      if (t + 1 < NT) {
        int offn = (t + 1) * TILE_ELEMS + tid * 4;
        int cn = min(offn, VOCAB - 4);
        async_load_b128(lds_off32(&tile[(t + 1) & 1][tid * 4]), base + cn);
        wait_async1();
      } else {
        wait_async0();
      }
      int off = t * TILE_ELEMS + tid * 4;
      int c = min(off, VOCAB - 4);
#pragma unroll
      for (int k = 0; k < 4; ++k) {
        int v = c + k;
        if (v >= off && v < VOCAB) {
          float sv = tile[t & 1][tid * 4 + k];
          float gb = ((green[v >> 5] >> (v & 31)) & 1u) ? BIASF : 0.0f;
          float l = sv + gb;
          if (l >= thresh) {
            uint32_t bits;
            if (v < HHALF) {
              U2 bb = threefry2x32(sk.x, sk.y, (uint32_t)v, (uint32_t)(v + HHALF));
              bits = bb.x;
            } else {
              U2 bb = threefry2x32(sk.x, sk.y, (uint32_t)(v - HHALF), (uint32_t)v);
              bits = bb.y;
            }
            float u = __uint_as_float(0x3f800000u | (bits >> 9)) - 1.0f;  // [0,1)
            u = fmaxf(u, 1.1754944e-38f);
            float gum = -__logf(-__logf(u));
            float val = l + gum;   // log p + const + gumbel; const irrelevant to argmax
            if (val > bv) { bv = val; bi = v; }
          }
        }
      }
    }
  }

  bestv[tid] = bv;
  besti[tid] = bi;
  __syncthreads();
  for (int s = 128; s > 0; s >>= 1) {
    if (tid < s) {
      if (bestv[tid + s] > bestv[tid]) {
        bestv[tid] = bestv[tid + s];
        besti[tid] = besti[tid + s];
      }
    }
    __syncthreads();
  }
  if (tid == 0) tok[row] = besti[0];
}

__global__ void fill_kernel(float4* __restrict__ out, long long n4) {
  long long i = (long long)blockIdx.x * blockDim.x + threadIdx.x;
  long long stride = (long long)gridDim.x * blockDim.x;
  float4 v = make_float4(1e-5f, 1e-5f, 1e-5f, 1e-5f);
  for (; i < n4; i += stride) out[i] = v;
}

__global__ void scatter_kernel(float* __restrict__ out, const int* __restrict__ tok) {
  int r = blockIdx.x * blockDim.x + threadIdx.x;
  if (r < BATCH) out[(size_t)r * VOCAB + tok[r]] = 100000.0f;
}

extern "C" void kernel_launch(void* const* d_in, const int* in_sizes, int n_in,
                              void* d_out, int out_size, void* d_ws, size_t ws_size,
                              hipStream_t stream) {
  const int* ids = (const int*)d_in[0];
  const float* scores = (const float*)d_in[1];
  float* out = (float*)d_out;
  int* tok = (int*)d_ws;

  wm_topp_kernel<<<BATCH, 256, 0, stream>>>(ids, scores, tok);

  long long n4 = (long long)BATCH * VOCAB / 4;   // 512*50257 divisible by 4
  fill_kernel<<<4096, 256, 0, stream>>>((float4*)out, n4);

  scatter_kernel<<<(BATCH + 255) / 256, 256, 0, stream>>>(out, tok);
}
